// MultiHeadSelfAttention_22479858827781
// MI455X (gfx1250) — matrix operand order repaired and checked by reading, not compiled, not run on hardware
//
#include <hip/hip_runtime.h>
#include <math.h>

// ---------------------------------------------------------------------------
// Multi-head self-attention with RoPE, causal mask — CDNA5 (gfx1250) WMMA path
// fp32 inputs -> bf16 WMMA (16x16x32, f32 accum) for all matrix math.
// B operands of the projection GEMMs staged to LDS via the Tensor Data Mover.
// ---------------------------------------------------------------------------

typedef __bf16 bf16;
typedef __attribute__((ext_vector_type(16))) __bf16 v16bf;
typedef __attribute__((ext_vector_type(4)))  __bf16 v4bf;
typedef __attribute__((ext_vector_type(8)))  float  v8f;
typedef __attribute__((ext_vector_type(4)))  float  v4f;
typedef __attribute__((ext_vector_type(4)))  unsigned int v4u;
typedef __attribute__((ext_vector_type(8)))  int    v8i;
typedef __attribute__((ext_vector_type(4)))  int    v4i;

constexpr int BATCH  = 2;
constexpr int SEQ    = 2048;
constexpr int DMODEL = 1024;
constexpr int HEADS  = 16;
constexpr int DK     = 64;
constexpr float LOG_THETA = 9.210340371976184f; // ln(10000)

#define WMMA_BF16(a, b, c) \
  __builtin_amdgcn_wmma_f32_16x16x32_bf16(false, (a), false, (b), (short)0, (c), false, false)

// ---- fragment loaders ------------------------------------------------------
// A fragment (16x32, 16-bit): lane L holds row M = L%16; the lane's 16
// elements are two contiguous 8-element K-chunks at kbase+(L/16)*8 and +16.
__device__ __forceinline__ v16bf load_frag_a(const bf16* base, int ld, int row0, int k0) {
  int lane = threadIdx.x & 31;
  int r  = lane & 15;
  int c0 = (lane >> 4) * 8;
  const bf16* p = base + (size_t)(row0 + r) * ld + k0 + c0;
  union { v16bf v; struct { v4f lo, hi; } q; } u;
  u.q.lo = *(const v4f*)(p);
  u.q.hi = *(const v4f*)(p + 16);
  return u.v;
}

// B fragment (32x16, 16-bit), operand stored N-major (row n contiguous in K):
// lane L holds column N = L%16; two 8-element K-chunks at (L/16)*8 and +16.
__device__ __forceinline__ v16bf load_frag_b(const bf16* base, int ld, int n0, int k0) {
  int lane = threadIdx.x & 31;
  int n  = lane & 15;
  int kk = (lane >> 4) * 8;
  const bf16* p = base + (size_t)(n0 + n) * ld + k0 + kk;
  union { v16bf v; struct { v4f lo, hi; } q; } u;
  u.q.lo = *(const v4f*)(p);
  u.q.hi = *(const v4f*)(p + 16);
  return u.v;
}

// ---- Tensor Data Mover: 2D bf16 tile Global -> LDS -------------------------
// D# packing per CDNA5 ISA ch.8 (group0: count/lds/global addr/type=2;
// group1: data_size=2B, tensor dims, tile dims, dim0 stride).
// 6-arg builtin form: (v4u, v8i, v4i, v4i, v8i, i32 cpol).
__device__ __forceinline__ void tdm_load_2d_bf16(unsigned lds_off, const void* gaddr,
                                                 unsigned tensor_d0, unsigned tensor_d1,
                                                 unsigned tile_d0, unsigned tile_d1,
                                                 unsigned long long d0_stride) {
  unsigned long long ga = (unsigned long long)gaddr;
  v4u g0;
  g0[0] = 1u;                                        // count=1, user mode
  g0[1] = lds_off;                                   // lds_addr (bytes)
  g0[2] = (unsigned)(ga & 0xFFFFFFFFu);              // global_addr[31:0]
  g0[3] = (unsigned)((ga >> 32) & 0x1FFFFFFu) | (2u << 30); // addr[56:32], type=2
  v8i g1;
  g1[0] = (int)(1u << 16);                           // data_size=1 -> 2 bytes
  g1[1] = (int)((tensor_d0 & 0xFFFFu) << 16);        // tensor_dim0[15:0] @bit48
  g1[2] = (int)(((tensor_d0 >> 16) & 0xFFFFu) | ((tensor_d1 & 0xFFFFu) << 16));
  g1[3] = (int)(((tensor_d1 >> 16) & 0xFFFFu) | ((tile_d0 & 0xFFFFu) << 16));
  g1[4] = (int)(tile_d1 & 0xFFFFu);                  // tile_dim1; tile_dim2=0
  g1[5] = (int)(d0_stride & 0xFFFFFFFFull);          // dim0_stride[31:0]
  g1[6] = (int)((d0_stride >> 32) & 0xFFFFull);      // dim0_stride[47:32]
  g1[7] = 0;                                         // dim1_stride unused (2D)
  v4i g2 = {0, 0, 0, 0};
  v4i g3 = {0, 0, 0, 0};
  v8i g4 = {0, 0, 0, 0, 0, 0, 0, 0};
  __builtin_amdgcn_tensor_load_to_lds(g0, g1, g2, g3, g4, 0);
}

// ---- elementwise fp32 -> bf16 (4-wide) ------------------------------------
__global__ __launch_bounds__(256) void cvt_f32_bf16_v4(const v4f* __restrict__ in,
                                                       v4bf* __restrict__ out, int n4) {
  int i = blockIdx.x * blockDim.x + threadIdx.x;
  if (i < n4) out[i] = __builtin_convertvector(in[i], v4bf);
}

// ---- GEMM: C[M,N] f32 = A_bf16[M,K] * B_bf16[N,K]^T ------------------------
// 256 threads = 8 waves. Block tile 256(M) x 64(N); wave w owns rows
// [blockIdx.y*256 + w*32, +32). B tile (64 x 256 K-chunk) staged to LDS by the
// TDM (issued by wave 0, TENSORcnt-tracked), A fragments streamed from global.
__global__ __launch_bounds__(256) void gemm_bf16_nt(const bf16* __restrict__ A,
                                                    const bf16* __restrict__ Bm,
                                                    float* __restrict__ C,
                                                    int M, int N, int K) {
  __shared__ __align__(16) bf16 ldsB[64 * 256];      // 32 KB
  int wave = threadIdx.x >> 5;
  int lane = threadIdx.x & 31;
  int row0 = blockIdx.y * 256 + wave * 32;
  int col0 = blockIdx.x * 64;
  v8f acc[2][4] = {};
  for (int ko = 0; ko < K; ko += 256) {
    if (wave == 0) {
      tdm_load_2d_bf16((unsigned)(unsigned long long)&ldsB[0],
                       Bm + (size_t)col0 * K + ko,
                       (unsigned)K, (unsigned)N, 256u, 64u,
                       (unsigned long long)K);
      __builtin_amdgcn_s_wait_tensorcnt(0);
    }
    __syncthreads();                                  // LDS tile ready
    if (ko + 256 < K)                                 // prefetch next A chunk
      __builtin_prefetch(A + (size_t)row0 * K + ko + 256, 0, 3);
#pragma unroll
    for (int kk = 0; kk < 256; kk += 32) {
      v16bf a0 = load_frag_a(A, K, row0, ko + kk);
      v16bf a1 = load_frag_a(A, K, row0 + 16, ko + kk);
#pragma unroll
      for (int t = 0; t < 4; ++t) {
        v16bf b = load_frag_b(ldsB, 256, t * 16, kk); // ds_load_b128 x2
        acc[0][t] = WMMA_BF16(a0, b, acc[0][t]);
        acc[1][t] = WMMA_BF16(a1, b, acc[1][t]);
      }
    }
    __syncthreads();                                  // done reading tile
  }
  int n = lane & 15, hgrp = lane >> 4;
#pragma unroll
  for (int i = 0; i < 2; ++i)
#pragma unroll
    for (int t = 0; t < 4; ++t)
#pragma unroll
      for (int e = 0; e < 8; ++e)
        C[(size_t)(row0 + i * 16 + e + 8 * hgrp) * N + col0 + t * 16 + n] =
            acc[i][t][e];
}

// ---- RoPE + head-split + bf16 (scale folds in 1/sqrt(dk) for Q) ------------
__global__ __launch_bounds__(256) void rope_kernel(const float* __restrict__ Xf,
                                                   bf16* __restrict__ Xb, float scale) {
  int i = blockIdx.x * blockDim.x + threadIdx.x;     // B*H*S*32 threads
  int p  = i & 31;
  int s  = (i >> 5) & (SEQ - 1);
  int bh = i >> 16;
  int b = bh >> 4, h = bh & 15;
  float inv = __expf(-LOG_THETA * (2.0f * p) / (float)DK);
  float ang = (float)s * inv;
  float c = __cosf(ang), sn = __sinf(ang);
  const float* src = Xf + ((size_t)(b * SEQ + s)) * DMODEL + h * DK + 2 * p;
  float xe = src[0], xo = src[1];
  bf16* dst = Xb + ((size_t)bh * SEQ + s) * DK + 2 * p;
  dst[0] = (bf16)((xe * c - xo * sn) * scale);
  dst[1] = (bf16)((xe * sn + xo * c) * scale);
}

// ---- V: (B,S,D) f32 -> (B,H,dk,S) bf16 -------------------------------------
__global__ __launch_bounds__(256) void vtrans_kernel(const float* __restrict__ Vf,
                                                     bf16* __restrict__ Vt) {
  int i = blockIdx.x * blockDim.x + threadIdx.x;     // B*H*DK*S threads
  int s  = i & (SEQ - 1);
  int d  = (i >> 11) & (DK - 1);
  int bh = i >> 17;
  int b = bh >> 4, h = bh & 15;
  Vt[((size_t)bh * DK + d) * SEQ + s] =
      (bf16)Vf[((size_t)(b * SEQ + s)) * DMODEL + h * DK + d];
}

// ---- Flash attention: causal, online softmax, bf16 WMMA --------------------
__global__ __launch_bounds__(128) void flash_attn_kernel(const bf16* __restrict__ Qb,
                                                         const bf16* __restrict__ Kb,
                                                         const bf16* __restrict__ Vt,
                                                         bf16* __restrict__ Ab) {
  __shared__ __align__(16) bf16 pbuf[4][16 * 32];    // per-wave P tile (16x32)
  int wave = threadIdx.x >> 5;
  int lane = threadIdx.x & 31;
  int bh = blockIdx.y;
  int b = bh >> 4, h = bh & 15;
  int q0 = blockIdx.x * 64 + wave * 16;

  const bf16* Qp = Qb + (size_t)bh * SEQ * DK;
  const bf16* Kp = Kb + (size_t)bh * SEQ * DK;
  const bf16* Vp = Vt + (size_t)bh * DK * SEQ;

  v16bf aQ0 = load_frag_a(Qp, DK, q0, 0);
  v16bf aQ1 = load_frag_a(Qp, DK, q0, 32);

  v8f o[4] = {};
  float mrow[8], lrow[8];
#pragma unroll
  for (int e = 0; e < 8; ++e) { mrow[e] = -1e30f; lrow[e] = 0.0f; }

  int n = lane & 15, hgrp = lane >> 4;
  int r = lane & 15, c0 = hgrp * 8;
  int kend = q0 + 16;                                // causal: k <= q0+15

  for (int kb = 0; kb < kend; kb += 32) {
    v8f sc[2] = {};
#pragma unroll
    for (int t = 0; t < 2; ++t) {
      v16bf bK0 = load_frag_b(Kp, DK, kb + t * 16, 0);
      sc[t] = WMMA_BF16(aQ0, bK0, sc[t]);
      v16bf bK1 = load_frag_b(Kp, DK, kb + t * 16, 32);
      sc[t] = WMMA_BF16(aQ1, bK1, sc[t]);
    }
#pragma unroll
    for (int e = 0; e < 8; ++e) {
      int qg = q0 + e + 8 * hgrp;
      float s0 = (kb + n      <= qg) ? sc[0][e] : -1e30f;
      float s1 = (kb + 16 + n <= qg) ? sc[1][e] : -1e30f;
      float mx = fmaxf(s0, s1);
#pragma unroll
      for (int off = 1; off < 16; off <<= 1)
        mx = fmaxf(mx, __shfl_xor(mx, off, 32));
      float mnew = fmaxf(mrow[e], mx);
      float corr = __expf(mrow[e] - mnew);
      float p0 = __expf(s0 - mnew);
      float p1 = __expf(s1 - mnew);
      float rs = p0 + p1;
#pragma unroll
      for (int off = 1; off < 16; off <<= 1)
        rs += __shfl_xor(rs, off, 32);
      lrow[e] = lrow[e] * corr + rs;
      mrow[e] = mnew;
#pragma unroll
      for (int t = 0; t < 4; ++t) o[t][e] *= corr;
      int m = e + 8 * hgrp;
      pbuf[wave][m * 32 + n]      = (bf16)p0;
      pbuf[wave][m * 32 + 16 + n] = (bf16)p1;
    }
    // C-fragment -> A-fragment through LDS (intra-wave in-order DS).
    asm volatile("s_wait_dscnt 0" ::: "memory");
    const bf16* pp = &pbuf[wave][r * 32 + c0];
    union { v16bf v; struct { v4f lo, hi; } q; } up;
    up.q.lo = *(const v4f*)(pp);
    up.q.hi = *(const v4f*)(pp + 16);
    v16bf aP = up.v;
#pragma unroll
    for (int t = 0; t < 4; ++t) {
      v16bf bV = load_frag_b(Vp, SEQ, t * 16, kb);
      o[t] = WMMA_BF16(aP, bV, o[t]);
    }
  }
#pragma unroll
  for (int t = 0; t < 4; ++t)
#pragma unroll
    for (int e = 0; e < 8; ++e) {
      float val = o[t][e] / (lrow[e] + 1e-9f);
      size_t idx = ((size_t)(b * SEQ + q0 + e + 8 * hgrp)) * DMODEL
                 + h * DK + t * 16 + n;
      Ab[idx] = (bf16)val;
    }
}

// ---------------------------------------------------------------------------
extern "C" void kernel_launch(void* const* d_in, const int* in_sizes, int n_in,
                              void* d_out, int out_size, void* d_ws, size_t ws_size,
                              hipStream_t stream) {
  const float* x  = (const float*)d_in[0];
  const float* wq = (const float*)d_in[1];
  const float* wk = (const float*)d_in[2];
  const float* wv = (const float*)d_in[3];
  const float* wo = (const float*)d_in[4];
  float* out = (float*)d_out;

  const size_t MB = 1ull << 20;
  char* ws = (char*)d_ws;
  bf16* xb  = (bf16*)(ws + 0 * MB);      // 4096x1024 bf16  (8 MB)
  bf16* wqb = (bf16*)(ws + 8 * MB);      // 1024x1024 bf16  (2 MB each)
  bf16* wkb = (bf16*)(ws + 10 * MB);
  bf16* wvb = (bf16*)(ws + 12 * MB);
  bf16* wob = (bf16*)(ws + 14 * MB);
  float* Qf = (float*)(ws + 16 * MB);    // 4096x1024 f32  (16 MB each)
  float* Kf = (float*)(ws + 32 * MB);
  float* Vf = (float*)(ws + 48 * MB);
  bf16* Qbh = (bf16*)(ws + 64 * MB);     // (B,H,S,dk) bf16 (8 MB each)
  bf16* Kbh = (bf16*)(ws + 72 * MB);
  bf16* Vth = (bf16*)(ws + 80 * MB);     // (B,H,dk,S)
  bf16* Ab  = (bf16*)(ws + 88 * MB);     // (B,S,D) bf16

  const int M = BATCH * SEQ;             // 4096
  const int N = DMODEL, K = DMODEL;

  // 1) fp32 -> bf16
  {
    int n4 = (M * DMODEL) / 4;
    cvt_f32_bf16_v4<<<(n4 + 255) / 256, 256, 0, stream>>>((const v4f*)x, (v4bf*)xb, n4);
    int w4 = (DMODEL * DMODEL) / 4;
    cvt_f32_bf16_v4<<<(w4 + 255) / 256, 256, 0, stream>>>((const v4f*)wq, (v4bf*)wqb, w4);
    cvt_f32_bf16_v4<<<(w4 + 255) / 256, 256, 0, stream>>>((const v4f*)wk, (v4bf*)wkb, w4);
    cvt_f32_bf16_v4<<<(w4 + 255) / 256, 256, 0, stream>>>((const v4f*)wv, (v4bf*)wvb, w4);
    cvt_f32_bf16_v4<<<(w4 + 255) / 256, 256, 0, stream>>>((const v4f*)wo, (v4bf*)wob, w4);
  }

  // 2) Q/K/V projections: (4096x1024) = xb * W^T   [WMMA bf16 + TDM staging]
  dim3 ggrid(N / 64, M / 256), gblk(256);
  gemm_bf16_nt<<<ggrid, gblk, 0, stream>>>(xb, wqb, Qf, M, N, K);
  gemm_bf16_nt<<<ggrid, gblk, 0, stream>>>(xb, wkb, Kf, M, N, K);
  gemm_bf16_nt<<<ggrid, gblk, 0, stream>>>(xb, wvb, Vf, M, N, K);

  // 3) RoPE (+1/sqrt(dk) folded into Q) and V transpose, all to bf16
  {
    int nr = BATCH * HEADS * SEQ * (DK / 2);
    rope_kernel<<<nr / 256, 256, 0, stream>>>(Qf, Qbh, 0.125f);
    rope_kernel<<<nr / 256, 256, 0, stream>>>(Kf, Kbh, 1.0f);
    int nv = BATCH * HEADS * DK * SEQ;
    vtrans_kernel<<<nv / 256, 256, 0, stream>>>(Vf, Vth);
  }

  // 4) causal flash attention [WMMA bf16]
  dim3 agrid(SEQ / 64, BATCH * HEADS), ablk(128);
  flash_attn_kernel<<<agrid, ablk, 0, stream>>>(Qbh, Kbh, Vth, Ab);

  // 5) output projection: out = attn * wo^T  (f32 result) [WMMA bf16]
  gemm_bf16_nt<<<ggrid, gblk, 0, stream>>>(Ab, wob, out, M, N, K);

  (void)in_sizes; (void)n_in; (void)out_size; (void)ws_size;
}
